// appnp_air_75213467287800
// MI455X (gfx1250) — matrix operand order repaired and checked by reading, not compiled, not run on hardware
//
#include <hip/hip_runtime.h>
#include <hip/hip_bf16.h>

typedef __attribute__((ext_vector_type(16))) __bf16 v16bf;
typedef __attribute__((ext_vector_type(8)))  float  v8f;

union FragBF { v16bf v; uint4 q[2]; };

// ---------------------------------------------------------------- constants
#define NN   100000
#define EE   1600000
#define FEAT 128
#define HID  256
#define CC   64
#define KHOP 10

// ---------------------------------------------------------------- fp32 -> bf16
__global__ void cvt_f32_bf16(const float* __restrict__ in, __bf16* __restrict__ out, long n) {
    long i = (long)blockIdx.x * blockDim.x + threadIdx.x;
    long stride = (long)gridDim.x * blockDim.x;
    for (; i < n; i += stride) out[i] = (__bf16)in[i];
}

// ---------------------------------------------------------------- pack weight [K,Nd] fp32 -> B-fragment bf16 order
// Fragment layout (wave32, 32x16 bf16 B tile, 8 dwords per lane):
//   lane l holds column n = nc*16 + (l&15); K = kc*32 + ((j>>2)<<4) + ((l>>4)<<3) + 2(j&3) + e
__global__ void pack_weight_bf16(const float* __restrict__ W, unsigned int* __restrict__ out,
                                 int K, int Nd) {
    int total = (K * Nd) >> 1;
    int ncCount = Nd >> 4;
    for (int idx = blockIdx.x * blockDim.x + threadIdx.x; idx < total;
         idx += gridDim.x * blockDim.x) {
        int j    = idx & 7;
        int lane = (idx >> 3) & 31;
        int rest = idx >> 8;
        int nc   = rest % ncCount;
        int kc   = rest / ncCount;
        int n = nc * 16 + (lane & 15);
        int k = kc * 32 + ((j >> 2) << 4) + ((lane >> 4) << 3) + 2 * (j & 3);
        __bf16 h0 = (__bf16)W[(long)k * Nd + n];
        __bf16 h1 = (__bf16)W[(long)(k + 1) * Nd + n];
        unsigned short u0 = __builtin_bit_cast(unsigned short, h0);
        unsigned short u1 = __builtin_bit_cast(unsigned short, h1);
        out[idx] = (unsigned int)u0 | ((unsigned int)u1 << 16);
    }
}

// ---------------------------------------------------------------- WMMA GEMM
// out = act( A[M,K](bf16) @ W[K,Nd] + bias (+ A residual) ).
// Wave tile: 32(M) x 64(N) -> 2 M-subtiles share each B fragment (8 WMMA / k-chunk).
template <int K, int Nd>
__global__ __launch_bounds__(256)
void gemm_wmma_bf16(const __bf16* __restrict__ A,
                    const uint4* __restrict__ Bp,      // packed fragments
                    const float* __restrict__ bias,
                    const float* __restrict__ preluA,  // nullptr -> no prelu
                    int M, int doResidual,
                    __bf16* __restrict__ outB,
                    float*  __restrict__ outF)
{
    constexpr int wavesN  = Nd / 64;        // waves along N (64 cols each)
    constexpr int wavesM  = 8 / wavesN;
    constexpr int ncCount = Nd / 16;
    constexpr int nIter   = K / 32;
    constexpr int bstride = ncCount * 1024; // packed-B bytes per k-chunk

    const int lane = threadIdx.x & 31;
    const int wave = threadIdx.x >> 5;
    const int wM = wave / wavesN;
    const int wN = wave % wavesN;
    const int rowBase = blockIdx.x * (32 * wavesM) + wM * 32;
    const int colBase = wN * 64;

    const int r0  = rowBase + (lane & 15);
    const int r1  = r0 + 16;
    const int rc0 = (r0 < M) ? r0 : (M - 1);
    const int rc1 = (r1 < M) ? r1 : (M - 1);
    const int koff = (lane >> 4) * 8;       // A-frag K offset per lane half

    const char* aptr0 = (const char*)(A + (size_t)rc0 * K + koff);
    const char* aptr1 = (const char*)(A + (size_t)rc1 * K + koff);
    const char* bptr  = (const char*)Bp + (size_t)lane * 32 + (size_t)(colBase >> 4) * 1024;

    v8f acc[2][4] = {};

#pragma unroll
    for (int it = 0; it < nIter; ++it) {
        FragBF a0, a1, b[4];
        a0.q[0] = *(const uint4*)(aptr0 + it * 64);
        a0.q[1] = *(const uint4*)(aptr0 + it * 64 + 32);
        a1.q[0] = *(const uint4*)(aptr1 + it * 64);
        a1.q[1] = *(const uint4*)(aptr1 + it * 64 + 32);
#pragma unroll
        for (int nt = 0; nt < 4; ++nt) {
            b[nt].q[0] = *(const uint4*)(bptr + (size_t)it * bstride + nt * 1024);
            b[nt].q[1] = *(const uint4*)(bptr + (size_t)it * bstride + nt * 1024 + 16);
        }
#pragma unroll
        for (int nt = 0; nt < 4; ++nt) {
            acc[0][nt] = __builtin_amdgcn_wmma_f32_16x16x32_bf16(
                false, a0.v, false, b[nt].v, (short)0, acc[0][nt], false, false);
            acc[1][nt] = __builtin_amdgcn_wmma_f32_16x16x32_bf16(
                false, a1.v, false, b[nt].v, (short)0, acc[1][nt], false, false);
        }
    }

    const int mOff   = (lane >> 4) * 8;     // C/D: vgpr i -> M = i + 8*(lane>=16)
    const int nLocal = lane & 15;
    const float alpha = preluA ? preluA[0] : 0.f;
    const int doPrelu = (preluA != nullptr);

#pragma unroll
    for (int m = 0; m < 2; ++m) {
#pragma unroll
        for (int nt = 0; nt < 4; ++nt) {
            const int col = colBase + nt * 16 + nLocal;
            const float bval = bias[col];
#pragma unroll
            for (int i = 0; i < 8; ++i) {
                const int row = rowBase + m * 16 + mOff + i;
                if (row < M) {
                    float v = acc[m][nt][i] + bval;
                    if (doResidual) v += (float)A[(size_t)row * K + col];   // K==Nd here
                    if (doPrelu)    v = (v >= 0.f) ? v : alpha * v;
                    if (outB) outB[(size_t)row * Nd + col] = (__bf16)v;
                    if (outF) outF[(size_t)row * Nd + col] = v;
                }
            }
        }
    }
}

// ---------------------------------------------------------------- zero fill
__global__ void zero_f32(float* __restrict__ p, long n) {
    long i = (long)blockIdx.x * blockDim.x + threadIdx.x;
    long stride = (long)gridDim.x * blockDim.x;
    for (; i < n; i += stride) p[i] = 0.f;
}

// ---------------------------------------------------------------- one propagation hop
// 16 threads per edge, float4 per thread; L2-resident gather + fp32 atomic scatter
__global__ void hop_kernel(const int* __restrict__ ei, const float* __restrict__ norm,
                           const float* __restrict__ hcur, float* __restrict__ hnxt) {
    long idx = (long)blockIdx.x * blockDim.x + threadIdx.x;
    long stride = (long)gridDim.x * blockDim.x;
    const long total = (long)EE * 16;
    for (; idx < total; idx += stride) {
        const int e = (int)(idx >> 4);
        const int t = (int)(idx & 15);
        const int src = ei[e];
        const int dst = ei[EE + e];
        const float w = norm[e];
        float4 v = *(const float4*)(hcur + (long)src * CC + t * 4);
        float* d = hnxt + (long)dst * CC + t * 4;
        __hip_atomic_fetch_add(d + 0, w * v.x, __ATOMIC_RELAXED, __HIP_MEMORY_SCOPE_AGENT);
        __hip_atomic_fetch_add(d + 1, w * v.y, __ATOMIC_RELAXED, __HIP_MEMORY_SCOPE_AGENT);
        __hip_atomic_fetch_add(d + 2, w * v.z, __ATOMIC_RELAXED, __HIP_MEMORY_SCOPE_AGENT);
        __hip_atomic_fetch_add(d + 3, w * v.w, __ATOMIC_RELAXED, __HIP_MEMORY_SCOPE_AGENT);
    }
}

// ---------------------------------------------------------------- hop retain score: sigmoid(pps . proj_w + b)
__global__ void score_kernel(const float* __restrict__ pps, const float* __restrict__ proj_w,
                             const float* __restrict__ proj_b, float* __restrict__ score,
                             int total) {   // total = (KHOP+1)*NN, flat idx = k*N+n
    int idx = blockIdx.x * blockDim.x + threadIdx.x;
    if (idx >= total) return;
    const float* row = pps + (long)idx * CC;
    float t = 0.f;
#pragma unroll
    for (int c = 0; c < CC; ++c) t += row[c] * proj_w[c];
    t += proj_b[0];
    score[idx] = 1.f / (1.f + __expf(-t));
}

// ---------------------------------------------------------------- combine hops + log_softmax (one wave32 per node)
__global__ void combine_logsoftmax(const float* __restrict__ pps, const float* __restrict__ score,
                                   float* __restrict__ out) {
    const int wid  = (int)(((long)blockIdx.x * blockDim.x + threadIdx.x) >> 5);
    const int lane = threadIdx.x & 31;
    if (wid >= NN) return;
    float a0 = 0.f, a1 = 0.f;
    for (int k = 0; k <= KHOP; ++k) {
        const float s = score[(long)k * NN + wid];
        const float* row = pps + ((long)k * NN + wid) * CC;
        a0 += s * row[lane];
        a1 += s * row[lane + 32];
    }
    float m = fmaxf(a0, a1);
#pragma unroll
    for (int off = 16; off > 0; off >>= 1) m = fmaxf(m, __shfl_xor(m, off, 32));
    float e = __expf(a0 - m) + __expf(a1 - m);
#pragma unroll
    for (int off = 16; off > 0; off >>= 1) e += __shfl_xor(e, off, 32);
    const float lse = m + __logf(e);
    out[(long)wid * CC + lane]      = a0 - lse;
    out[(long)wid * CC + lane + 32] = a1 - lse;
}

// ---------------------------------------------------------------- launcher
extern "C" void kernel_launch(void* const* d_in, const int* in_sizes, int n_in,
                              void* d_out, int out_size, void* d_ws, size_t ws_size,
                              hipStream_t stream) {
    const float* x        = (const float*)d_in[0];
    const int*   ei       = (const int*)  d_in[1];
    const float* norm     = (const float*)d_in[2];
    const float* W1 = (const float*)d_in[3];  const float* b1 = (const float*)d_in[4];
    const float* W2 = (const float*)d_in[5];  const float* b2 = (const float*)d_in[6];
    const float* W3 = (const float*)d_in[7];  const float* b3 = (const float*)d_in[8];
    const float* W4 = (const float*)d_in[9];  const float* b4 = (const float*)d_in[10];
    const float* prelu_a = (const float*)d_in[11];
    const float* proj_w  = (const float*)d_in[12];
    const float* proj_b  = (const float*)d_in[13];
    float* out = (float*)d_out;

    // ---- carve workspace (256B aligned)
    char* base = (char*)d_ws;
    size_t off = 0;
    auto carve = [&](size_t bytes) -> void* {
        void* p = base + off;
        off = (off + bytes + 255) & ~(size_t)255;
        return p;
    };
    __bf16* Xb  = (__bf16*)carve((size_t)NN * FEAT * 2);
    __bf16* H1b = (__bf16*)carve((size_t)NN * HID * 2);
    __bf16* H2b = (__bf16*)carve((size_t)NN * HID * 2);
    unsigned int* W1p = (unsigned int*)carve((size_t)FEAT * HID * 2);
    unsigned int* W2p = (unsigned int*)carve((size_t)HID * HID * 2);
    unsigned int* W3p = (unsigned int*)carve((size_t)HID * HID * 2);
    unsigned int* W4p = (unsigned int*)carve((size_t)HID * CC * 2);
    float* pps   = (float*)carve((size_t)(KHOP + 1) * NN * CC * 4);
    float* score = (float*)carve((size_t)(KHOP + 1) * NN * 4);

    // ---- 1. activations/weights to bf16 fragment-friendly forms
    cvt_f32_bf16<<<2048, 256, 0, stream>>>(x, Xb, (long)NN * FEAT);
    pack_weight_bf16<<<64,  256, 0, stream>>>(W1, W1p, FEAT, HID);
    pack_weight_bf16<<<128, 256, 0, stream>>>(W2, W2p, HID, HID);
    pack_weight_bf16<<<128, 256, 0, stream>>>(W3, W3p, HID, HID);
    pack_weight_bf16<<<32,  256, 0, stream>>>(W4, W4p, HID, CC);

    // ---- 2. MLP via WMMA  (blockM = 64 for Nd=256, 256 for Nd=64)
    const int gridH = (NN + 63) / 64;     // 1563
    const int gridC = (NN + 255) / 256;   // 391
    gemm_wmma_bf16<FEAT, HID><<<gridH, 256, 0, stream>>>(
        Xb, (const uint4*)W1p, b1, prelu_a, NN, 0, H1b, nullptr);
    gemm_wmma_bf16<HID, HID><<<gridH, 256, 0, stream>>>(
        H1b, (const uint4*)W2p, b2, prelu_a, NN, 1, H2b, nullptr);
    gemm_wmma_bf16<HID, HID><<<gridH, 256, 0, stream>>>(
        H2b, (const uint4*)W3p, b3, prelu_a, NN, 1, H1b, nullptr);
    gemm_wmma_bf16<HID, CC><<<gridC, 256, 0, stream>>>(
        H1b, (const uint4*)W4p, b4, nullptr, NN, 0, nullptr, pps /* z = pps[0] */);

    // ---- 3. K hops (L2-resident gather + atomic scatter)
    for (int k = 0; k < KHOP; ++k) {
        float* hc = pps + (long)k * NN * CC;
        float* hn = hc + (long)NN * CC;
        zero_f32<<<2048, 256, 0, stream>>>(hn, (long)NN * CC);
        hop_kernel<<<50000, 256, 0, stream>>>(ei, norm, hc, hn);
    }

    // ---- 4. per-hop retain scores
    const int totalKN = (KHOP + 1) * NN;
    score_kernel<<<(totalKN + 255) / 256, 256, 0, stream>>>(pps, proj_w, proj_b, score, totalKN);

    // ---- 5. adaptive combine + log_softmax
    combine_logsoftmax<<<(NN * 32 + 255) / 256, 256, 0, stream>>>(pps, score, out);
    (void)in_sizes; (void)n_in; (void)out_size; (void)ws_size;
}
